// IRRETORE_21852793602449
// MI455X (gfx1250) — matrix operand order repaired and checked by reading, not compiled, not run on hardware
//
#include <hip/hip_runtime.h>
#include <math.h>

// Problem constants (match reference)
#define Bb   2
#define Pp   16384
#define Nn   4096
#define KNB  16
#define CFc  64

typedef __attribute__((ext_vector_type(16))) _Float16 v16h;
typedef __attribute__((ext_vector_type(8)))  _Float16 v8h;
typedef __attribute__((ext_vector_type(8)))  float    v8f;

static __device__ inline v8f vzero8f() {
  v8f z;
#pragma unroll
  for (int i = 0; i < 8; ++i) z[i] = 0.f;
  return z;
}

// ---------------------------------------------------------------------------
// Kernel 1: brute-force KNN (16 nearest of 4096, 2-D). dst cloud cached in LDS.
// ---------------------------------------------------------------------------
__global__ void __launch_bounds__(256) knn_kernel(const float* __restrict__ src,
                                                  const float* __restrict__ dst,
                                                  int* __restrict__ idx) {
  __shared__ float sxy[2 * Nn];  // 32 KB
  const int b = blockIdx.y;
  const int tid = threadIdx.x;
  for (int i = tid; i < 2 * Nn; i += 256) sxy[i] = dst[(size_t)b * Nn * 2 + i];
  __syncthreads();

  const int p = blockIdx.x * 256 + tid;
  const float sx = src[((size_t)b * Pp + p) * 2 + 0];
  const float sy = src[((size_t)b * Pp + p) * 2 + 1];

  float bd[KNB];
  int   bi[KNB];
#pragma unroll
  for (int j = 0; j < KNB; ++j) { bd[j] = 3.4e38f; bi[j] = 0; }

  for (int i = 0; i < Nn; ++i) {
    const float dx = sxy[2 * i] - sx;
    const float dy = sxy[2 * i + 1] - sy;
    const float d = dx * dx + dy * dy;
    if (d < bd[KNB - 1]) {
      bd[KNB - 1] = d; bi[KNB - 1] = i;
#pragma unroll
      for (int j = KNB - 1; j > 0; --j) {
        if (bd[j] < bd[j - 1]) {
          const float td = bd[j]; bd[j] = bd[j - 1]; bd[j - 1] = td;
          const int   ti = bi[j]; bi[j] = bi[j - 1]; bi[j - 1] = ti;
        }
      }
    }
  }
#pragma unroll
  for (int j = 0; j < KNB; ++j) idx[((size_t)b * Pp + p) * KNB + j] = bi[j];
}

// ---------------------------------------------------------------------------
// Kernel 2: pack weights [O][C] f32 -> [KP][O] f16 (transposed, K zero-padded)
// so a B-fragment is 16 contiguous f16 at row k (lane = N within tile).
// ---------------------------------------------------------------------------
__global__ void pack_w_kernel(const float* __restrict__ src, _Float16* __restrict__ dst,
                              int O, int C, int KP) {
  const int t = blockIdx.x * 256 + threadIdx.x;
  if (t >= KP * O) return;
  const int k = t / O, o = t % O;
  dst[t] = (k < C) ? (_Float16)src[o * C + k] : (_Float16)0.f;
}

// ---------------------------------------------------------------------------
// WMMA fragment loaders (wave32, v_wmma_f32_16x16x32_f16 layouts)
// ---------------------------------------------------------------------------
// A (16xK f16 tile, row-major in LDS, 16B-aligned rows):
//   lane L: row m = L&15; elems 0..7 = K {kc + h*8 ..}, elems 8..15 = K {kc+16+h*8 ..}
static __device__ inline v16h load_a_frag(const _Float16* base, int stride, int kc, int lane) {
  const int m = lane & 15, h = (lane >> 4) & 1;
  const _Float16* p = base + m * stride + kc + h * 8;
  v8h lo = *(const v8h*)p;
  v8h hi = *(const v8h*)(p + 16);
  v16h a;
#pragma unroll
  for (int e = 0; e < 8; ++e) { a[e] = lo[e]; a[e + 8] = hi[e]; }
  return a;
}
// B (K x N f16, packed [K][ncols] in global): lane L: col n = nt*16+(L&15); elems 0..7 = K {kc + h*8 ..}, elems 8..15 = K {kc+16+h*8 ..}
static __device__ inline v16h load_b_frag(const _Float16* __restrict__ w, int ncols,
                                          int kc, int nt, int lane) {
  const int n = nt * 16 + (lane & 15), h = (lane >> 4) & 1;
  const _Float16* p = w + (size_t)(kc + h * 8) * ncols + n;
  v16h b;
#pragma unroll
  for (int e = 0; e < 8; ++e) { b[e] = p[(size_t)e * ncols]; b[e + 8] = p[(size_t)(16 + e) * ncols]; }
  return b;
}

// Conv(1x1) -> BN(affine) -> ReLU layer: D = relu((A*W + b)*s + t), D re-laid to LDS f16.
template <int KDIM, int NOUT>
static __device__ inline void cbr_layer(const _Float16* in, int instride,
                                        const _Float16* __restrict__ W,
                                        const float* __restrict__ bias,
                                        const float* __restrict__ sc,
                                        const float* __restrict__ tb,
                                        _Float16* out, int outstride, int lane) {
  constexpr int NCH = KDIM / 32;
  constexpr int NT  = NOUT / 16;
  v16h a[NCH];
#pragma unroll
  for (int c = 0; c < NCH; ++c) a[c] = load_a_frag(in, instride, c * 32, lane);
  v8f acc[NT];
#pragma unroll
  for (int t = 0; t < NT; ++t) {
    acc[t] = vzero8f();
#pragma unroll
    for (int c = 0; c < NCH; ++c) {
      v16h bf = load_b_frag(W, NOUT, c * 32, t, lane);
      acc[t] = __builtin_amdgcn_wmma_f32_16x16x32_f16(false, a[c], false, bf,
                                                      (short)0, acc[t], false, false);
    }
  }
  // D layout: lane L holds N = t*16 + (L&15), M = 8*(L>>4) + r
  const int nlo = lane & 15, mb = 8 * ((lane >> 4) & 1);
#pragma unroll
  for (int t = 0; t < NT; ++t) {
    const int n = t * 16 + nlo;
    const float bn = bias[n], sn = sc[n], tn = tb[n];
#pragma unroll
    for (int r = 0; r < 8; ++r) {
      const float v = (acc[t][r] + bn) * sn + tn;
      out[(mb + r) * outstride + n] = (_Float16)(v > 0.f ? v : 0.f);
    }
  }
  __syncthreads();
}

// ---------------------------------------------------------------------------
// Kernel 3: fused gather + filter MLP + color MLP. One wave = one query point,
// its 16 neighbors are the M dimension of every WMMA.
// ---------------------------------------------------------------------------
__global__ void __launch_bounds__(256) fused_mlp_kernel(
    const float* __restrict__ rc, const float* __restrict__ xy,
    const float* __restrict__ value, const float* __restrict__ grad,
    const float* __restrict__ feat, const int* __restrict__ idx,
    const _Float16* __restrict__ Wt0, const _Float16* __restrict__ Wt1,
    const _Float16* __restrict__ Wt2, const _Float16* __restrict__ Wtm,
    const _Float16* __restrict__ Wc0, const _Float16* __restrict__ Wc1,
    const float* __restrict__ b0, const float* __restrict__ s0, const float* __restrict__ t0,
    const float* __restrict__ b1, const float* __restrict__ s1, const float* __restrict__ t1,
    const float* __restrict__ b2, const float* __restrict__ s2, const float* __restrict__ t2,
    const float* __restrict__ bm, const float* __restrict__ sbn2, const float* __restrict__ tbn2,
    const float* __restrict__ bc0, const float* __restrict__ sc0, const float* __restrict__ tc0,
    const float* __restrict__ bc1, const float* __restrict__ sc1, const float* __restrict__ tc1,
    const float* __restrict__ wcm, const float* __restrict__ bcm,
    float* __restrict__ out) {
  __shared__ _Float16 nf[8][16][96];   // 24 KB: per-wave neighbor-feature tile
  __shared__ _Float16 act[8][16][64];  // 16 KB: per-wave activation tile

  const int lane = threadIdx.x & 31;
  const int wv   = threadIdx.x >> 5;
  const int b    = blockIdx.y;
  const int p    = blockIdx.x * 8 + wv;
  _Float16* NF  = &nf[wv][0][0];
  _Float16* ACT = &act[wv][0][0];
  const int m = lane & 15, h = (lane >> 4) & 1;

  // ---- gather & build nf tile (channels: cxy2,gxy2,gval3,ggrad6,feat64,rel2,dist1,pad16)
  const int nb = idx[((size_t)b * Pp + p) * KNB + m];
  const float* frow = feat + ((size_t)b * Nn + nb) * CFc;
  _Float16* row = NF + m * 96;
  if (h == 0) {
    const float sx = rc[((size_t)b * Pp + p) * 2 + 0];
    const float sy = rc[((size_t)b * Pp + p) * 2 + 1];
    const float gx = xy[((size_t)b * Nn + nb) * 2 + 0];
    const float gy = xy[((size_t)b * Nn + nb) * 2 + 1];
    row[0] = (_Float16)sx; row[1] = (_Float16)sy;
    row[2] = (_Float16)gx; row[3] = (_Float16)gy;
#pragma unroll
    for (int j = 0; j < 3; ++j) row[4 + j] = (_Float16)value[((size_t)b * Nn + nb) * 3 + j];
#pragma unroll
    for (int j = 0; j < 6; ++j) row[7 + j] = (_Float16)grad[((size_t)b * Nn + nb) * 6 + j];
    const float rx = gx - sx, ry = gy - sy;
    row[77] = (_Float16)rx; row[78] = (_Float16)ry;
    row[79] = (_Float16)sqrtf(rx * rx + ry * ry);
#pragma unroll
    for (int j = 0; j < 32; ++j) row[13 + j] = (_Float16)frow[j];
  } else {
#pragma unroll
    for (int j = 0; j < 32; ++j) row[45 + j] = (_Float16)frow[32 + j];
#pragma unroll
    for (int j = 0; j < 16; ++j) row[80 + j] = (_Float16)0.f;
  }
  __syncthreads();

  // ---- filter branch: 80->64->64->64 (cbr) ----
  cbr_layer<96, 64>(NF, 96, Wt0, b0, s0, t0, ACT, 64, lane);
  cbr_layer<64, 64>(ACT, 64, Wt1, b1, s1, t1, ACT, 64, lane);
  cbr_layer<64, 64>(ACT, 64, Wt2, b2, s2, t2, ACT, 64, lane);

  // map2 (plain linear) + relu(gfeat*f*s+t) + mean over the 16 neighbors (M dim)
  {
    const v16h a0 = load_a_frag(ACT, 64, 0, lane);
    const v16h a1 = load_a_frag(ACT, 64, 32, lane);
    const int nlo = lane & 15, mb = 8 * h;
#pragma unroll
    for (int t = 0; t < 4; ++t) {
      v8f acc = vzero8f();
      acc = __builtin_amdgcn_wmma_f32_16x16x32_f16(false, a0, false,
              load_b_frag(Wtm, 64, 0, t, lane), (short)0, acc, false, false);
      acc = __builtin_amdgcn_wmma_f32_16x16x32_f16(false, a1, false,
              load_b_frag(Wtm, 64, 32, t, lane), (short)0, acc, false, false);
      const int n = t * 16 + nlo;
      const float bmn = bm[n], sn = sbn2[n], tn = tbn2[n];
      float partial = 0.f;
#pragma unroll
      for (int r = 0; r < 8; ++r) {
        const float f  = acc[r] + bmn;
        const float gf = (float)NF[(mb + r) * 96 + 13 + n];
        const float hh = gf * f * sn + tn;
        partial += (hh > 0.f ? hh : 0.f);
      }
      const float total = partial + __shfl_xor(partial, 16, 32);
      if (lane < 16) out[((size_t)b * 67 + 3 + n) * Pp + p] = total * (1.f / 16.f);
    }
  }
  __syncthreads();

  // ---- color branch: 80->64->32 (cbr) ----
  cbr_layer<96, 64>(NF, 96, Wc0, bc0, sc0, tc0, ACT, 64, lane);
  cbr_layer<64, 32>(ACT, 64, Wc1, bc1, sc1, tc1, ACT, 64, lane);

  // color_map2 (32->3) + softmax over neighbors + weighted sum of group_value
  {
    float cj0 = bcm[0], cj1 = bcm[1], cj2 = bcm[2];
#pragma unroll
    for (int k = 0; k < 32; ++k) {
      const float av = (float)ACT[m * 64 + k];
      cj0 += av * wcm[k];
      cj1 += av * wcm[32 + k];
      cj2 += av * wcm[64 + k];
    }
    const float cj[3] = {cj0, cj1, cj2};
#pragma unroll
    for (int j = 0; j < 3; ++j) {
      const float v = cj[j];
      float mx = v;
#pragma unroll
      for (int off = 8; off >= 1; off >>= 1) mx = fmaxf(mx, __shfl_xor(mx, off, 32));
      const float e = __expf(v - mx);
      float s = e;
      float w = e * (float)NF[m * 96 + 4 + j];
#pragma unroll
      for (int off = 8; off >= 1; off >>= 1) {
        s += __shfl_xor(s, off, 32);
        w += __shfl_xor(w, off, 32);
      }
      if (lane == 0) out[((size_t)b * 67 + j) * Pp + p] = w / s;
    }
  }
}

// ---------------------------------------------------------------------------
extern "C" void kernel_launch(void* const* d_in, const int* in_sizes, int n_in,
                              void* d_out, int out_size, void* d_ws, size_t ws_size,
                              hipStream_t stream) {
  (void)in_sizes; (void)n_in; (void)out_size; (void)ws_size;
  const float* rc   = (const float*)d_in[0];
  const float* xy   = (const float*)d_in[1];
  const float* val  = (const float*)d_in[2];
  const float* grad = (const float*)d_in[3];
  const float* feat = (const float*)d_in[4];
  const float* w0 = (const float*)d_in[5];
  const float* b0 = (const float*)d_in[6];
  const float* s0 = (const float*)d_in[7];
  const float* t0 = (const float*)d_in[8];
  const float* w1 = (const float*)d_in[9];
  const float* b1 = (const float*)d_in[10];
  const float* s1 = (const float*)d_in[11];
  const float* t1 = (const float*)d_in[12];
  const float* w2 = (const float*)d_in[13];
  const float* b2 = (const float*)d_in[14];
  const float* s2 = (const float*)d_in[15];
  const float* t2 = (const float*)d_in[16];
  const float* wm = (const float*)d_in[17];
  const float* bm = (const float*)d_in[18];
  const float* sbn2 = (const float*)d_in[19];
  const float* tbn2 = (const float*)d_in[20];
  const float* wc0 = (const float*)d_in[21];
  const float* bc0 = (const float*)d_in[22];
  const float* sc0 = (const float*)d_in[23];
  const float* tc0 = (const float*)d_in[24];
  const float* wc1 = (const float*)d_in[25];
  const float* bc1 = (const float*)d_in[26];
  const float* sc1 = (const float*)d_in[27];
  const float* tc1 = (const float*)d_in[28];
  const float* wcm = (const float*)d_in[29];
  const float* bcm = (const float*)d_in[30];
  float* out = (float*)d_out;

  // workspace layout: [idx B*P*K int32][packed f16 weights]
  int* idx = (int*)d_ws;
  const size_t idx_bytes = (size_t)Bb * Pp * KNB * sizeof(int);
  _Float16* wp  = (_Float16*)((char*)d_ws + idx_bytes);
  _Float16* Wt0 = wp;
  _Float16* Wt1 = Wt0 + 96 * 64;
  _Float16* Wt2 = Wt1 + 64 * 64;
  _Float16* Wtm = Wt2 + 64 * 64;
  _Float16* Wc0 = Wtm + 64 * 64;
  _Float16* Wc1 = Wc0 + 96 * 64;

  pack_w_kernel<<<(96 * 64 + 255) / 256, 256, 0, stream>>>(w0,  Wt0, 64, 80, 96);
  pack_w_kernel<<<(64 * 64 + 255) / 256, 256, 0, stream>>>(w1,  Wt1, 64, 64, 64);
  pack_w_kernel<<<(64 * 64 + 255) / 256, 256, 0, stream>>>(w2,  Wt2, 64, 64, 64);
  pack_w_kernel<<<(64 * 64 + 255) / 256, 256, 0, stream>>>(wm,  Wtm, 64, 64, 64);
  pack_w_kernel<<<(96 * 64 + 255) / 256, 256, 0, stream>>>(wc0, Wc0, 64, 80, 96);
  pack_w_kernel<<<(64 * 32 + 255) / 256, 256, 0, stream>>>(wc1, Wc1, 32, 64, 64);

  knn_kernel<<<dim3(Pp / 256, Bb), 256, 0, stream>>>(rc, xy, idx);

  fused_mlp_kernel<<<dim3(Pp / 8, Bb), 256, 0, stream>>>(
      rc, xy, val, grad, feat, idx,
      Wt0, Wt1, Wt2, Wtm, Wc0, Wc1,
      b0, s0, t0, b1, s1, t1, b2, s2, t2, bm, sbn2, tbn2,
      bc0, sc0, tc0, bc1, sc1, tc1, wcm, bcm, out);
}
